// PointnetSAModuleMSGVotes_4209067950528
// MI455X (gfx1250) — matrix operand order repaired and checked by reading, not compiled, not run on hardware
//
#include <hip/hip_runtime.h>
#include <stdint.h>
#include <stddef.h>

typedef __bf16 bf16;
typedef __attribute__((ext_vector_type(16))) __bf16 v16bf;
typedef __attribute__((ext_vector_type(8)))  __bf16 v8bf;
typedef __attribute__((ext_vector_type(8)))  float  v8f;

#define NB   4
#define NPTS 16384
#define NS   1024     // npoint (S)
#define NC   256      // feature channels
#define KPAD 288      // 259 padded up to multiple of 32
#define NSEG 16       // ball-query scan segments
#define SEGLEN (NPTS / NSEG)

// ---------------- workspace layout (bytes) ----------------
#define OFF_NEWXYZ 0ull                                   // B*S*3 f32       = 49152
#define OFF_FEATT  49152ull                               // B*N*256 bf16    = 33554432
#define OFF_IDX0   (OFF_FEATT + 33554432ull)              // B*S*32 i32      = 524288
#define OFF_IDX1   (OFF_IDX0 + 524288ull)                 // B*S*64 i32      = 1048576
#define OFF_WB     (OFF_IDX1 + 1048576ull)                // 256*288 bf16    = 147456
#define OFF_STATS  (OFF_WB + 147456ull)                   // 512 f32         = 2048
#define OFF_AB     (OFF_STATS + 2048ull)                  // scale/shift     = 4096
#define OFF_SIDX0  (OFF_AB + 4096ull)                     // B*S*16*32 i32   = 8388608
#define OFF_SIDX1  (OFF_SIDX0 + 8388608ull)               // B*S*16*64 i32   = 16777216
#define OFF_SCNT0  (OFF_SIDX1 + 16777216ull)              // B*S*16 i32      = 262144
#define OFF_SCNT1  (OFF_SCNT0 + 262144ull)                // B*S*16 i32      = 262144
#define OFF_X      (OFF_SCNT1 + 262144ull)                // 262144*288 bf16 = 150994944
#define OFF_Y      (OFF_X + 150994944ull)                 // 262144*256 bf16 = 134217728

// ---------------- kernel 1: new_xyz gather + inds passthrough ----------------
__global__ void k_newxyz(const float* __restrict__ xyz, const int* __restrict__ inds,
                         float* __restrict__ nx_ws, float* __restrict__ out_xyz,
                         int* __restrict__ out_inds) {
  int i = blockIdx.x * blockDim.x + threadIdx.x;
  if (i >= NB * NS) return;
  int b = i / NS;
  int n = inds[i];
  const float* s = xyz + ((size_t)b * NPTS + n) * 3;
  float x = s[0], y = s[1], z = s[2];
  nx_ws[(size_t)i * 3 + 0] = x; nx_ws[(size_t)i * 3 + 1] = y; nx_ws[(size_t)i * 3 + 2] = z;
  out_xyz[(size_t)i * 3 + 0] = x; out_xyz[(size_t)i * 3 + 1] = y; out_xyz[(size_t)i * 3 + 2] = z;
  out_inds[i] = n;
}

// ---------------- kernel 2: features (B,C,N) -> point-major bf16 (B,N,C) ----------------
__global__ void k_feat_transpose(const float* __restrict__ feat, bf16* __restrict__ featT) {
  __shared__ float tile[32][33];
  int b = blockIdx.z;
  int n0 = blockIdx.x * 32;
  int c0 = blockIdx.y * 32;
  int tx = threadIdx.x, ty = threadIdx.y;
  for (int j = ty; j < 32; j += 8)
    tile[j][tx] = feat[((size_t)b * NC + (c0 + j)) * NPTS + (n0 + tx)];
  __syncthreads();
  for (int j = ty; j < 32; j += 8)
    featT[((size_t)b * NPTS + (n0 + j)) * NC + (c0 + tx)] = (bf16)tile[tx][j];
}

// ---------------- kernel 3a: segmented dual-radius ball scan ----------------
// block 256 = 16 s (lanes, uniform xyz address per half-wave) x 16 segments
__global__ __launch_bounds__(256)
void k_bq_scan(const float* __restrict__ xyz, const float* __restrict__ newxyz,
               int* __restrict__ sidx0, int* __restrict__ sidx1,
               int* __restrict__ scnt0, int* __restrict__ scnt1) {
  int b = blockIdx.y;
  int sl = threadIdx.x & 15;
  int seg = threadIdx.x >> 4;
  int s = blockIdx.x * 16 + sl;
  size_t bs = (size_t)b * NS + s;
  float cx = newxyz[bs * 3 + 0], cy = newxyz[bs * 3 + 1], cz = newxyz[bs * 3 + 2];
  size_t gi = bs * NSEG + seg;
  int* o0 = sidx0 + gi * 32;
  int* o1 = sidx1 + gi * 64;
  int cnt0 = 0, cnt1 = 0;
  const float* xb = xyz + (size_t)b * NPTS * 3;
  int n0 = seg * SEGLEN;
  for (int j = 0; j < SEGLEN; ++j) {
    int n = n0 + j;
    float dx = xb[n * 3 + 0] - cx;
    float dy = xb[n * 3 + 1] - cy;
    float dz = xb[n * 3 + 2] - cz;
    float d2 = dx * dx + dy * dy + dz * dz;
    if (d2 < 0.64f) {                       // 0.8^2
      if (cnt1 < 64) o1[cnt1++] = n;
      if (d2 < 0.16f && cnt0 < 32) o0[cnt0++] = n;   // 0.4^2
      if (cnt1 == 64 && cnt0 == 32) break;  // segment can't contribute more
    }
  }
  scnt0[gi] = cnt0;
  scnt1[gi] = cnt1;
}

// ---------------- kernel 3b: ordered merge of segments ----------------
__global__ void k_bq_merge(const int* __restrict__ sidx0, const int* __restrict__ sidx1,
                           const int* __restrict__ scnt0, const int* __restrict__ scnt1,
                           int* __restrict__ idx0, int* __restrict__ idx1) {
  int i = blockIdx.x * blockDim.x + threadIdx.x;   // (b,s) flat
  if (i >= NB * NS) return;
  size_t bs = (size_t)i;
  int* o1 = idx1 + bs * 64;
  int cnt1 = 0;
  for (int seg = 0; seg < NSEG && cnt1 < 64; ++seg) {
    size_t gi = bs * NSEG + seg;
    int c = scnt1[gi];
    const int* src = sidx1 + gi * 64;
    for (int k = 0; k < c && cnt1 < 64; ++k) o1[cnt1++] = src[k];
  }
  int p1 = cnt1 ? o1[0] : 0;
  for (int k = cnt1; k < 64; ++k) o1[k] = p1;

  int* o0 = idx0 + bs * 32;
  int cnt0 = 0;
  for (int seg = 0; seg < NSEG && cnt0 < 32; ++seg) {
    size_t gi = bs * NSEG + seg;
    int c = scnt0[gi];
    const int* src = sidx0 + gi * 32;
    for (int k = 0; k < c && cnt0 < 32; ++k) o0[cnt0++] = src[k];
  }
  int p0 = cnt0 ? o0[0] : 0;
  for (int k = cnt0; k < 32; ++k) o0[k] = p0;
}

// ---------------- kernel 4: group gather -> X (P, KPAD) bf16, vectorized ----------------
// channel order: [feat 0..255][xyz 256..258][pad 259..287] (W columns permuted to match)
__global__ __launch_bounds__(256)
void k_gather(const bf16* __restrict__ featT, const float* __restrict__ xyz,
              const float* __restrict__ newxyz, const int* __restrict__ idx,
              bf16* __restrict__ X, int ns) {
  int b = blockIdx.y, s = blockIdx.x;
  size_t bs = (size_t)b * NS + s;
  int nchunk = ns * (KPAD / 8);                  // 36 chunks of 8 bf16 per row
  for (int cc = threadIdx.x; cc < nchunk; cc += 256) {
    int j = cc / 36, c8 = cc - j * 36;
    int n = idx[bs * ns + j];
    size_t p = bs * ns + j;
    v8bf* dst = (v8bf*)(X + p * KPAD) + c8;
    if (c8 < 32) {
      *dst = *((const v8bf*)(featT + ((size_t)b * NPTS + n) * NC) + c8);
    } else {
      v8bf v;
#pragma unroll
      for (int e = 0; e < 8; ++e) v[e] = (bf16)0.0f;
      if (c8 == 32) {
#pragma unroll
        for (int e = 0; e < 3; ++e)
          v[e] = (bf16)(xyz[((size_t)b * NPTS + n) * 3 + e] - newxyz[bs * 3 + e]);
      }
      *dst = v;
    }
  }
}

// ---------------- kernel 5: weight f32 -> padded bf16 (optionally column-permuted) ----------------
__global__ void k_wconv(const float* __restrict__ Wf, bf16* __restrict__ Wb,
                        int M, int Kin, int Kp, int perm) {
  int i = blockIdx.x * 256 + threadIdx.x;
  if (i >= M * Kp) return;
  int m = i / Kp, k = i % Kp;
  if (perm) {           // layer 0: X channel k -> reference input channel
    if (k < 259) {
      int ksrc = (k < 256) ? (k + 3) : (k - 256);
      Wb[i] = (bf16)Wf[(size_t)m * 259 + ksrc];
    } else {
      Wb[i] = (bf16)0.0f;
    }
  } else {
    Wb[i] = (k < Kin) ? (bf16)Wf[(size_t)m * Kin + k] : (bf16)0.0f;
  }
}

// ---------------- kernel 6: WMMA GEMM  Y(P,M) = W(M,Kp) * X(P,Kp)^T ----------------
// block = 256 threads = 8 waves; block tile 64(M) x 128(P); wave tile 32x32
__global__ __launch_bounds__(256)
void k_gemm_bf16(const bf16* __restrict__ W, const bf16* __restrict__ X,
                 bf16* __restrict__ Y, int M, int Kp, long P) {
  __shared__ __align__(16) bf16 sW[64 * KPAD];
  int tid = threadIdx.x;
  int lane = tid & 31, wave = tid >> 5;
  int waveM = wave >> 2, waveN = wave & 3;
  int mblock = blockIdx.y * 64;
  long pblock = (long)blockIdx.x * 128;

  // stage the 64-row W tile into LDS (16B chunks)
  int nch = (64 * Kp) / 8;
  for (int c = tid; c < nch; c += 256) {
    int flat = c * 8;
    int row = flat / Kp, kk = flat % Kp;
    *(v8bf*)&sW[row * Kp + kk] = *(const v8bf*)&W[(size_t)(mblock + row) * Kp + kk];
  }
  __syncthreads();

  v8f zero = {0.f, 0.f, 0.f, 0.f, 0.f, 0.f, 0.f, 0.f};
  v8f acc[2][2];
  acc[0][0] = zero; acc[0][1] = zero; acc[1][0] = zero; acc[1][1] = zero;

  int m0 = waveM * 32;                // LDS-local row base
  long p0 = pblock + waveN * 32;
  int g = lane >> 4;                  // lane half
  int lr = lane & 15;

  for (int k0 = 0; k0 < Kp; k0 += 32) {
    __builtin_prefetch(&X[(size_t)(p0 + lr) * Kp + k0 + 32], 0, 1);

    // A fragments (16-bit A layout: lane<16 -> K{0..7,16..23}, lane>=16 -> +8)
    v16bf a[2];
#pragma unroll
    for (int mi = 0; mi < 2; ++mi) {
      const bf16* base = &sW[(size_t)(m0 + mi * 16 + lr) * Kp + k0 + 8 * g];
      v8bf lo = *(const v8bf*)base;
      v8bf hi = *(const v8bf*)(base + 16);
      a[mi] = __builtin_shufflevector(lo, hi, 0, 1, 2, 3, 4, 5, 6, 7,
                                      8, 9, 10, 11, 12, 13, 14, 15);
    }
    // B fragments (same K order as A: lane<16 -> K{0..7,16..23}, lane>=16 -> +8)
    v16bf bf_[2];
#pragma unroll
    for (int ni = 0; ni < 2; ++ni) {
      const bf16* base = &X[(size_t)(p0 + ni * 16 + lr) * Kp + k0 + 8 * g];
      v8bf lo = *(const v8bf*)base;
      v8bf hi = *(const v8bf*)(base + 16);
      bf_[ni] = __builtin_shufflevector(lo, hi, 0, 1, 2, 3, 4, 5, 6, 7,
                                        8, 9, 10, 11, 12, 13, 14, 15);
    }
#pragma unroll
    for (int mi = 0; mi < 2; ++mi)
#pragma unroll
      for (int ni = 0; ni < 2; ++ni)
        acc[mi][ni] = __builtin_amdgcn_wmma_f32_16x16x32_bf16(
            false, a[mi], false, bf_[ni], (short)0, acc[mi][ni], false, false);
  }

  // store: D layout -> lane<16: M=r, lane>=16: M=8+r; col = lane&15
  int mhalf = g * 8;
#pragma unroll
  for (int mi = 0; mi < 2; ++mi)
#pragma unroll
    for (int ni = 0; ni < 2; ++ni) {
      long p = p0 + ni * 16 + lr;
      int m = mblock + m0 + mi * 16 + mhalf;
      v8bf o;
#pragma unroll
      for (int r = 0; r < 8; ++r) o[r] = (bf16)acc[mi][ni][r];
      *(v8bf*)&Y[(size_t)p * M + m] = o;
    }
}

// ---------------- kernel 7: BN stats ----------------
__global__ void k_statzero(float* stats) { stats[threadIdx.x] = 0.f; }

__global__ __launch_bounds__(256)
void k_stataccum(const bf16* __restrict__ Y, float* __restrict__ stats, int M, long P) {
  __shared__ float r1[256], r2[256];
  int tid = threadIdx.x;
  int ch = tid & 63, pr = tid >> 6;
  int m = blockIdx.x * 64 + ch;
  long chunk = P / gridDim.y;
  long pb = (long)blockIdx.y * chunk;
  float s1 = 0.f, s2 = 0.f;
  for (long p = pb + pr; p < pb + chunk; p += 4) {
    float v = (float)Y[(size_t)p * M + m];
    s1 += v; s2 += v * v;
  }
  r1[tid] = s1; r2[tid] = s2;
  __syncthreads();
  if (pr == 0) {
    float t1 = r1[ch] + r1[ch + 64] + r1[ch + 128] + r1[ch + 192];
    float t2 = r2[ch] + r2[ch + 64] + r2[ch + 128] + r2[ch + 192];
    atomicAdd(&stats[m], t1);
    atomicAdd(&stats[M + m], t2);
  }
}

// scale -> ab[m], shift -> ab[256+m]
__global__ void k_statfinal(const float* __restrict__ stats,
                            const float* __restrict__ gamma, const float* __restrict__ beta,
                            float* __restrict__ ab, int M, float invP) {
  int m = threadIdx.x;
  if (m >= M) return;
  float mean = stats[m] * invP;
  float var = stats[M + m] * invP - mean * mean;
  float sc = gamma[m] * rsqrtf(var + 1e-5f);
  ab[m] = sc;
  ab[256 + m] = beta[m] - mean * sc;
}

// ---------------- kernel 8: fused BN scale/shift + ReLU, 8-wide (bf16 in place) ----------------
__global__ __launch_bounds__(256)
void k_bnrelu(bf16* __restrict__ Y, const float* __restrict__ ab, long total8, int M) {
  long i = (long)blockIdx.x * 256 + threadIdx.x;
  if (i >= total8) return;
  size_t base = (size_t)i * 8;
  int m0 = (int)(base & (size_t)(M - 1));     // M power of two, 8 | base, m0+7 < M
  v8bf v = *(v8bf*)(Y + base);
  v8bf o;
#pragma unroll
  for (int e = 0; e < 8; ++e) {
    float f = (float)v[e] * ab[m0 + e] + ab[256 + m0 + e];
    o[e] = (bf16)(f > 0.f ? f : 0.f);
  }
  *(v8bf*)(Y + base) = o;
}

// ---------------- kernel 9: fused BN+ReLU + max-pool, LDS-transposed coalesced output ----------------
// block = 256 channels; each block covers 16 s-values; out (B,512,S)
__global__ __launch_bounds__(256)
void k_maxpool(const bf16* __restrict__ Y, const float* __restrict__ ab,
               float* __restrict__ out, int ns, int co_base) {
  __shared__ float tile[256][17];
  int b = blockIdx.y;
  int c = threadIdx.x;
  float sc = ab[c], sh = ab[256 + c];
  for (int e = 0; e < 16; ++e) {
    int s = blockIdx.x * 16 + e;
    size_t bs = (size_t)b * NS + s;
    const bf16* src = Y + bs * (size_t)ns * 256 + c;
    float mv = -1e30f;
    for (int j = 0; j < ns; ++j) {
      float v = (float)src[(size_t)j * 256] * sc + sh;
      v = v > 0.f ? v : 0.f;
      mv = v > mv ? v : mv;
    }
    tile[c][e] = mv;
  }
  __syncthreads();
  float4* op = (float4*)(out + ((size_t)b * 512 + co_base + c) * NS + (size_t)blockIdx.x * 16);
#pragma unroll
  for (int q = 0; q < 4; ++q) {
    float4 v = {tile[c][q * 4 + 0], tile[c][q * 4 + 1], tile[c][q * 4 + 2], tile[c][q * 4 + 3]};
    op[q] = v;
  }
}

// ---------------- host launcher ----------------
extern "C" void kernel_launch(void* const* d_in, const int* in_sizes, int n_in,
                              void* d_out, int out_size, void* d_ws, size_t ws_size,
                              hipStream_t stream) {
  const float* xyz      = (const float*)d_in[0];
  const float* features = (const float*)d_in[1];
  const int*   inds     = (const int*)d_in[2];

  uint8_t* w8 = (uint8_t*)d_ws;
  float* nx_ws  = (float*)(w8 + OFF_NEWXYZ);
  bf16*  featT  = (bf16*) (w8 + OFF_FEATT);
  int*   idx0   = (int*)  (w8 + OFF_IDX0);
  int*   idx1   = (int*)  (w8 + OFF_IDX1);
  bf16*  Wb     = (bf16*) (w8 + OFF_WB);
  float* stats  = (float*)(w8 + OFF_STATS);
  float* ab     = (float*)(w8 + OFF_AB);
  int*   sidx0  = (int*)  (w8 + OFF_SIDX0);
  int*   sidx1  = (int*)  (w8 + OFF_SIDX1);
  int*   scnt0  = (int*)  (w8 + OFF_SCNT0);
  int*   scnt1  = (int*)  (w8 + OFF_SCNT1);
  bf16*  Xbuf   = (bf16*) (w8 + OFF_X);
  bf16*  Ybuf   = (bf16*) (w8 + OFF_Y);

  float* out      = (float*)d_out;
  float* out_xyz  = out;                           // B*S*3 = 12288
  float* out_feat = out + 12288;                   // B*512*S = 2097152
  int*   out_inds = (int*)(out + 12288 + 2097152); // B*S = 4096

  // 1) new_xyz + inds
  k_newxyz<<<(NB * NS + 255) / 256, 256, 0, stream>>>(xyz, inds, nx_ws, out_xyz, out_inds);
  // 2) transpose features to point-major bf16
  k_feat_transpose<<<dim3(NPTS / 32, NC / 32, NB), dim3(32, 8), 0, stream>>>(features, featT);
  // 3) segmented ball query: parallel scan + ordered merge
  k_bq_scan<<<dim3(NS / 16, NB), 256, 0, stream>>>(xyz, nx_ws, sidx0, sidx1, scnt0, scnt1);
  k_bq_merge<<<(NB * NS + 255) / 256, 256, 0, stream>>>(sidx0, sidx1, scnt0, scnt1, idx0, idx1);

  const int layerM[3]   = {128, 128, 256};
  const int layerKp[3]  = {KPAD, 128, 128};
  const int layerKin[3] = {259, 128, 128};

  for (int si = 0; si < 2; ++si) {
    int ns = si ? 64 : 32;
    const int* idx = si ? idx1 : idx0;
    long P = (long)NB * NS * ns;

    // 4) gather into Xbuf (feat-first channel order)
    k_gather<<<dim3(NS, NB), 256, 0, stream>>>(featT, xyz, nx_ws, idx, Xbuf, ns);

    bf16* curX = Xbuf;
    bf16* curY = Ybuf;
    for (int li = 0; li < 3; ++li) {
      int base = 3 + (si * 3 + li) * 3;           // pytree order: beta, gamma, w
      const float* beta  = (const float*)d_in[base + 0];
      const float* gamma = (const float*)d_in[base + 1];
      const float* wgt   = (const float*)d_in[base + 2];
      int M = layerM[li], Kp = layerKp[li], Kin = layerKin[li];

      k_wconv<<<(M * Kp + 255) / 256, 256, 0, stream>>>(wgt, Wb, M, Kin, Kp, li == 0 ? 1 : 0);
      k_gemm_bf16<<<dim3((unsigned)(P / 128), M / 64), 256, 0, stream>>>(Wb, curX, curY, M, Kp, P);
      k_statzero<<<1, 512, 0, stream>>>(stats);
      k_stataccum<<<dim3(M / 64, 512), 256, 0, stream>>>(curY, stats, M, P);
      k_statfinal<<<1, 256, 0, stream>>>(stats, gamma, beta, ab, M, 1.0f / (float)P);
      if (li < 2) {  // last layer's BN+ReLU is fused into the max-pool
        long total8 = P * (long)M / 8;
        k_bnrelu<<<(unsigned)((total8 + 255) / 256), 256, 0, stream>>>(curY, ab, total8, M);
      }
      bf16* t = curX; curX = curY; curY = t;      // output becomes next input
    }
    // after 3 swaps the last layer's raw output is in curX; ab holds layer-3 scale/shift
    k_maxpool<<<dim3(NS / 16, NB), 256, 0, stream>>>(curX, ab, out_feat, ns, si * 256);
  }
}
